// BNN_2blk_50_20_47201690583462
// MI455X (gfx1250) — matrix operand order repaired and checked by reading, not compiled, not run on hardware
//
#include <hip/hip_runtime.h>
#include <cstdint>

typedef __bf16 bf16_t;
typedef __attribute__((ext_vector_type(16))) __bf16 v16bf;
typedef __attribute__((ext_vector_type(8)))  __bf16 v8bf;
typedef __attribute__((ext_vector_type(8)))  float  v8f;

#define EPS_BN 1e-4f
#define B_TOTAL 65536
#define D_IN 1024
#define H1 50
#define H1P 64
#define H2 20
#define NOUT 10
#define ROWS_PER_BLOCK 128
#define THREADS 256
#define KSTEPS (D_IN / 32)
#define WSTRIDE 40   /* bf16 elems per staged-weight row (80B, 16B aligned) */
#define H1STRIDE 66  /* floats per h1 row (bank-stagger) */

__global__ __launch_bounds__(THREADS) void bnn_fused(
    const float* __restrict__ x,
    const float* __restrict__ w1, const float* __restrict__ b1,
    const float* __restrict__ g1, const float* __restrict__ be1,
    const float* __restrict__ m1, const float* __restrict__ v1,
    const float* __restrict__ w2, const float* __restrict__ b2,
    const float* __restrict__ g2, const float* __restrict__ be2,
    const float* __restrict__ m2, const float* __restrict__ v2,
    const float* __restrict__ w5, const float* __restrict__ b5,
    float* __restrict__ out)
{
  __shared__ __align__(16) bf16_t sW[2][H1P * WSTRIDE];
  __shared__ float   sH1[ROWS_PER_BLOCK * H1STRIDE];
  __shared__ float   sA1[H1], sC1[H1];
  __shared__ float   sA2[H2], sC2[H2];
  __shared__ uint32_t sW2lo[H2], sW2hi[H2];
  __shared__ uint32_t sW5b[NOUT];
  __shared__ float   sB5[NOUT];

  const int t    = threadIdx.x;
  const int wave = t >> 5;
  const int lane = t & 31;
  const int half = lane >> 4;
  const int l16  = lane & 15;

  // ---- per-block precompute: BN affine folds + binarized w2/w5 bitmasks ----
  if (t < H1) {
    float A = g1[t] * rsqrtf(v1[t] + EPS_BN);
    sA1[t] = A;
    sC1[t] = A * (b1[t] - m1[t]) + be1[t];
  }
  if (t < H2) {
    float A = g2[t] * rsqrtf(v2[t] + EPS_BN);
    sA2[t] = A;
    sC2[t] = A * (b2[t] - m2[t]) + be2[t];
    uint32_t lo = 0u, hi = 0u;
    for (int j = 0; j < 32; ++j) lo |= (w2[t * H1 + j] >= 0.f ? 1u : 0u) << j;
    for (int j = 32; j < H1; ++j) hi |= (w2[t * H1 + j] >= 0.f ? 1u : 0u) << (j - 32);
    sW2lo[t] = lo; sW2hi[t] = hi;
  }
  if (t < NOUT) {
    uint32_t b = 0u;
    for (int j = 0; j < H2; ++j) b |= (w5[t * H2 + j] >= 0.f ? 1u : 0u) << j;
    sW5b[t] = b; sB5[t] = b5[t];
  }

  // ---- cooperative stager: sign(w1) K-slab -> bf16 ±1 in LDS ----
  auto stage = [&](int ks, int buf) {
    int n  = t >> 2;                    // 0..63 (output neuron, padded)
    int kb = (t & 3) << 3;              // 0,8,16,24
    int srow = (n < H1) ? n : (H1 - 1); // clamp padding rows (never consumed)
    const float4* p =
        reinterpret_cast<const float4*>(w1 + (size_t)srow * D_IN + ks * 32 + kb);
    float4 a = p[0];
    float4 c = p[1];
    uint32_t u0 = ((a.x >= 0.f) ? 0x3F80u : 0xBF80u) | (((a.y >= 0.f) ? 0x3F80u : 0xBF80u) << 16);
    uint32_t u1 = ((a.z >= 0.f) ? 0x3F80u : 0xBF80u) | (((a.w >= 0.f) ? 0x3F80u : 0xBF80u) << 16);
    uint32_t u2 = ((c.x >= 0.f) ? 0x3F80u : 0xBF80u) | (((c.y >= 0.f) ? 0x3F80u : 0xBF80u) << 16);
    uint32_t u3 = ((c.z >= 0.f) ? 0x3F80u : 0xBF80u) | (((c.w >= 0.f) ? 0x3F80u : 0xBF80u) << 16);
    uint4 pk; pk.x = u0; pk.y = u1; pk.z = u2; pk.w = u3;
    *reinterpret_cast<uint4*>(&sW[buf][n * WSTRIDE + kb]) = pk;
  };

  stage(0, 0);
  __syncthreads();

  // each wave: 16 rows x 64 cols; lane row fixed for entire K loop
  const int myRow = blockIdx.x * ROWS_PER_BLOCK + wave * 16 + l16;
  const float* xrow = x + (size_t)myRow * D_IN + half * 8;

  v8f acc[4] = {};

  for (int ks = 0; ks < KSTEPS; ++ks) {
    const int buf = ks & 1;
    if (ks + 1 < KSTEPS) stage(ks + 1, buf ^ 1);

    // A fragment: two contiguous 8-float segments (lane l + lane l+16 cover
    // exactly one 128B line per row per step -> perfect streaming)
    const float4* px = reinterpret_cast<const float4*>(xrow + ks * 32);
    float4 xa = px[0], xb = px[1];   // k = 8h + 0..7
    float4 xc = px[4], xd = px[5];   // k = 8h + 16..23
    float xf[16] = {xa.x, xa.y, xa.z, xa.w, xb.x, xb.y, xb.z, xb.w,
                    xc.x, xc.y, xc.z, xc.w, xd.x, xd.y, xd.z, xd.w};
    v16bf ahi, alo;  // fp32 = hi + lo bf16 split (near-fp32 accuracy)
    #pragma unroll
    for (int e = 0; e < 16; ++e) {
      float f = xf[e];
      __bf16 h = (__bf16)f;
      ahi[e] = h;
      alo[e] = (__bf16)(f - (float)h);
    }

    #pragma unroll
    for (int ti = 0; ti < 4; ++ti) {
      // B fragment: lane half h holds K=8h..8h+7 then K=16+8h..16+8h+7, col = l16
      const bf16_t* wp = &sW[buf][(ti * 16 + l16) * WSTRIDE + half * 8];
      v8bf b0 = *reinterpret_cast<const v8bf*>(wp);
      v8bf b1v = *reinterpret_cast<const v8bf*>(wp + 16);
      v16bf bb = __builtin_shufflevector(b0, b1v, 0, 1, 2, 3, 4, 5, 6, 7,
                                         8, 9, 10, 11, 12, 13, 14, 15);
      acc[ti] = __builtin_amdgcn_wmma_f32_16x16x32_bf16(
          false, alo, false, bb, (short)0, acc[ti], false, false);
      acc[ti] = __builtin_amdgcn_wmma_f32_16x16x32_bf16(
          false, ahi, false, bb, (short)0, acc[ti], false, false);
    }
    __syncthreads();
  }

  // spill fc1 accumulators to LDS (C layout: VGPR r -> M = r + 8*half, N = l16)
  #pragma unroll
  for (int ti = 0; ti < 4; ++ti) {
    #pragma unroll
    for (int r = 0; r < 8; ++r) {
      int m = r + half * 8;
      sH1[(wave * 16 + m) * H1STRIDE + ti * 16 + l16] = acc[ti][r];
    }
  }
  __syncthreads();

  // ---- tail: BN1+sign, XNOR-popcount fc2, BN2+sign, fc5, log_softmax ----
  if (t < ROWS_PER_BLOCK) {
    const float* hrow = &sH1[t * H1STRIDE];
    uint32_t lo = 0u, hi = 0u;
    #pragma unroll
    for (int j = 0; j < 32; ++j)
      lo |= ((sA1[j] * hrow[j] + sC1[j]) >= 0.f ? 1u : 0u) << j;
    #pragma unroll
    for (int j = 32; j < H1; ++j)
      hi |= ((sA1[j] * hrow[j] + sC1[j]) >= 0.f ? 1u : 0u) << (j - 32);

    uint32_t s2 = 0u;
    #pragma unroll
    for (int o = 0; o < H2; ++o) {
      int neg = __popc(lo ^ sW2lo[o]) + __popc(hi ^ sW2hi[o]);
      float d2 = (float)(H1 - 2 * neg);             // exact ±1 dot product
      s2 |= ((sA2[o] * d2 + sC2[o]) >= 0.f ? 1u : 0u) << o;
    }

    float lg[NOUT];
    float mx = -3.4e38f;
    #pragma unroll
    for (int o = 0; o < NOUT; ++o) {
      int neg = __popc(s2 ^ sW5b[o]);
      lg[o] = (float)(H2 - 2 * neg) + sB5[o];
      mx = fmaxf(mx, lg[o]);
    }
    float ssum = 0.f;
    #pragma unroll
    for (int o = 0; o < NOUT; ++o) ssum += expf(lg[o] - mx);
    float lse = logf(ssum);
    size_t grow = (size_t)blockIdx.x * ROWS_PER_BLOCK + t;
    #pragma unroll
    for (int o = 0; o < NOUT; ++o) out[grow * NOUT + o] = lg[o] - mx - lse;
  }
}

extern "C" void kernel_launch(void* const* d_in, const int* in_sizes, int n_in,
                              void* d_out, int out_size, void* d_ws, size_t ws_size,
                              hipStream_t stream) {
  (void)in_sizes; (void)n_in; (void)out_size; (void)d_ws; (void)ws_size;
  const float* x   = (const float*)d_in[0];
  const float* w1  = (const float*)d_in[1];
  const float* b1  = (const float*)d_in[2];
  const float* g1  = (const float*)d_in[3];
  const float* be1 = (const float*)d_in[4];
  const float* m1  = (const float*)d_in[5];
  const float* v1  = (const float*)d_in[6];
  const float* w2  = (const float*)d_in[7];
  const float* b2  = (const float*)d_in[8];
  const float* g2  = (const float*)d_in[9];
  const float* be2 = (const float*)d_in[10];
  const float* m2  = (const float*)d_in[11];
  const float* v2  = (const float*)d_in[12];
  const float* w5  = (const float*)d_in[13];
  const float* b5  = (const float*)d_in[14];
  float* out = (float*)d_out;

  dim3 grid(B_TOTAL / ROWS_PER_BLOCK);  // 512 blocks
  dim3 block(THREADS);                  // 8 waves (wave32)
  bnn_fused<<<grid, block, 0, stream>>>(x, w1, b1, g1, be1, m1, v1,
                                        w2, b2, g2, be2, m2, v2, w5, b5, out);
}
